// MHA_2516850835986
// MI455X (gfx1250) — matrix operand order repaired and checked by reading, not compiled, not run on hardware
//
#include <hip/hip_runtime.h>

typedef __bf16 bf16;
typedef __bf16 v16bf __attribute__((ext_vector_type(16)));
typedef __bf16 v4bf  __attribute__((ext_vector_type(4)));
typedef float  v8f   __attribute__((ext_vector_type(8)));

#define SLDA 2052   // 16-row score strip row stride (8*SLDA % 64 != 0 -> no bank clash)

// ---------------- WMMA helpers (CDNA5 gfx1250, wave32) ----------------

__device__ __forceinline__ v8f wmma_bf16(v16bf a, v16bf b, v8f c) {
  // (neg_a, A, neg_b, B, c_mod, C, reuse_a, reuse_b)
  return __builtin_amdgcn_wmma_f32_16x16x32_bf16(false, a, false, b, (short)0, c, false, false);
}

// A fragment: 16x32 bf16, row-major source. ISA 7.12.2:
// lanes 0-15: M=lane, K = {0..7} U {16..23}; lanes 16-31: M=lane-16, K = {8..15} U {24..31}
__device__ __forceinline__ v16bf load_a_frag(const bf16* __restrict__ base, int ld, int m0, int k0) {
  int lane = threadIdx.x & 31;
  const bf16* p = base + (m0 + (lane & 15)) * ld + k0 + ((lane >> 4) << 3);
  union { uint4 u[2]; v16bf v; } t;
  t.u[0] = *reinterpret_cast<const uint4*>(p);        // K +0..7
  t.u[1] = *reinterpret_cast<const uint4*>(p + 16);   // K +16..23
  return t.v;
}

// B fragment: 32x16 bf16 from an N-major ("B^T row-major") source: row n, contiguous k.
// lanes 0-15: N=lane, K = {0..7} U {16..23}; lanes 16-31: N=lane-16, K = {8..15} U {24..31} (2 K per VGPR).
__device__ __forceinline__ v16bf load_b_frag(const bf16* __restrict__ base, int ld, int n0, int k0) {
  int lane = threadIdx.x & 31;
  const bf16* p = base + (n0 + (lane & 15)) * ld + k0 + ((lane >> 4) << 3);
  union { uint4 u[2]; v16bf v; } t;
  t.u[0] = *reinterpret_cast<const uint4*>(p);        // K +0..7
  t.u[1] = *reinterpret_cast<const uint4*>(p + 16);   // K +16..23
  return t.v;
}

// ---------------- prep kernels ----------------

__global__ __launch_bounds__(256) void cvt_f32_bf16x4(const float4* __restrict__ in,
                                                      v4bf* __restrict__ out, int n4) {
  int i = blockIdx.x * 256 + threadIdx.x;
  if (i < n4) {
    float4 f = in[i];
    v4bf o;
    o[0] = (bf16)f.x; o[1] = (bf16)f.y; o[2] = (bf16)f.z; o[3] = (bf16)f.w;
    out[i] = o;
  }
}

// in: [K][N] f32 row-major -> out: [N][K] bf16 row-major (LDS-tiled 32x32)
__global__ __launch_bounds__(256) void transpose_cvt(const float* __restrict__ in,
                                                     bf16* __restrict__ out, int K, int N) {
  __shared__ float tile[32][33];
  int n0 = blockIdx.x * 32, k0 = blockIdx.y * 32;
  int tx = threadIdx.x, ty = threadIdx.y;           // (32, 8)
#pragma unroll
  for (int i = 0; i < 4; ++i)
    tile[ty + 8 * i][tx] = in[(k0 + ty + 8 * i) * N + (n0 + tx)];
  __syncthreads();
#pragma unroll
  for (int i = 0; i < 4; ++i)
    out[(n0 + ty + 8 * i) * K + (k0 + tx)] = (bf16)tile[tx][ty + 8 * i];
}

// ---------------- GEMM1: qkv = x @ w_qkv, fused RoPE, scatter to Q/K/V^T ----------------
// Each wave: 64 rows x 64 cols (4x4 register tile, 16 WMMAs / K-step -> 32 FLOP/B from L2).
// wT is w_qkv transposed [6144][2048] bf16. Q,K stored [h][t][64] bf16 (post-RoPE);
// V stored transposed [h][64][t] bf16. Each 64-col strip is exactly one head.
__global__ __launch_bounds__(256) void gemm_qkv_rope(const bf16* __restrict__ xb,
                                                     const bf16* __restrict__ wT,
                                                     bf16* __restrict__ Qb,
                                                     bf16* __restrict__ Kb,
                                                     bf16* __restrict__ Vt) {
  int wid = blockIdx.x * 8 + (threadIdx.x >> 5);
  int m0 = (wid / 96) * 64;
  int n0 = (wid % 96) * 64;
  v8f acc[4][4] = {};
  for (int kb = 0; kb < 2048; kb += 32) {
    v16bf a[4];
#pragma unroll
    for (int mi = 0; mi < 4; ++mi) a[mi] = load_a_frag(xb, 2048, m0 + 16 * mi, kb);
#pragma unroll
    for (int j = 0; j < 4; ++j) {
      v16bf b = load_b_frag(wT, 2048, n0 + 16 * j, kb);
#pragma unroll
      for (int mi = 0; mi < 4; ++mi) acc[mi][j] = wmma_bf16(a[mi], b, acc[mi][j]);
    }
  }
  int lane = threadIdx.x & 31;
  int rb = (lane >> 4) * 8;
  int region = n0 >> 11;                 // 0=q, 1=k, 2=v (wave-uniform)
  int h = (n0 & 2047) >> 6;
  // Reference quirk: rotation angle indexed by HEAD (H == d/2 broadcast): theta^(-h/32)
  float invf = __powf(10000.0f, -(float)h * (1.0f / 32.0f));
#pragma unroll
  for (int mi = 0; mi < 4; ++mi) {
    int mbase = m0 + 16 * mi + rb;
    if (region == 2) {
#pragma unroll
      for (int j = 0; j < 4; ++j) {
        int dd = 16 * j + (lane & 15);
        bf16* dst = Vt + (h * 64 + dd) * 2048 + mbase;
#pragma unroll
        for (int r = 0; r < 8; ++r) dst[r] = (bf16)acc[mi][j][r];
      }
    } else {
      float sn[8], cn[8];
#pragma unroll
      for (int r = 0; r < 8; ++r)
        __sincosf((float)(mbase + r) * invf, &sn[r], &cn[r]);
      bf16* dstb = (region == 0 ? Qb : Kb) + (h * 2048 + mbase) * 64;
#pragma unroll
      for (int j = 0; j < 4; ++j) {
        int dd = 16 * j + (lane & 15);
#pragma unroll
        for (int r = 0; r < 8; ++r) {
          float v = acc[mi][j][r];
          float o = __shfl_xor(v, 1, 32);  // RoPE partner: adjacent column = adjacent lane
          float res = ((dd & 1) == 0) ? (v * cn[r] - o * sn[r]) : (o * sn[r] + v * cn[r]);
          dstb[r * 64 + dd] = (bf16)res;
        }
      }
    }
  }
}

// ---------------- attention: per (head, 16-query block), score strip in LDS ----------------
__global__ __launch_bounds__(256) void attn(const bf16* __restrict__ Qb,
                                            const bf16* __restrict__ Kb,
                                            const bf16* __restrict__ Vt,
                                            const float* __restrict__ mask,
                                            bf16* __restrict__ Yb) {
  extern __shared__ float smem[];
  float* S = smem;                     // [16][SLDA]
  float* Oacc = smem + 16 * SLDA;      // [16][64]
  int h = blockIdx.x >> 7;
  int q0 = (blockIdx.x & 127) << 4;
  int tid = threadIdx.x, wave = tid >> 5, lane = tid & 31;
  int rb = (lane >> 4) * 8;

  for (int i = tid; i < 1024; i += 256) Oacc[i] = 0.0f;

  const bf16* Qh = Qb + (h * 2048 + q0) * 64;
  const bf16* Kh = Kb + h * 2048 * 64;
  const bf16* Vh = Vt + h * 64 * 2048;

  // ---- S = Q K^T / 8 + mask : wave owns 256 key columns ----
  v16bf qa0 = load_a_frag(Qh, 64, 0, 0);
  v16bf qa1 = load_a_frag(Qh, 64, 0, 32);
  int cw = wave * 256;
  for (int ct = 0; ct < 16; ++ct) {
    int n0 = cw + ct * 16;
    v8f s8 = {};
    s8 = wmma_bf16(qa0, load_b_frag(Kh, 64, n0, 0), s8);
    s8 = wmma_bf16(qa1, load_b_frag(Kh, 64, n0, 32), s8);
    int col = n0 + (lane & 15);
#pragma unroll
    for (int r = 0; r < 8; ++r) {
      int row = q0 + rb + r;
      S[(rb + r) * SLDA + col] = s8[r] * 0.125f + mask[row * 2048 + col];
    }
  }
  __syncthreads();

  // ---- softmax: each wave handles 2 rows with wave32 shfl reductions ----
#pragma unroll
  for (int rr = 0; rr < 2; ++rr) {
    float* Srow = S + (wave * 2 + rr) * SLDA;
    float mx = -3.0e38f;
    for (int c = lane; c < 2048; c += 32) mx = fmaxf(mx, Srow[c]);
#pragma unroll
    for (int off = 16; off; off >>= 1) mx = fmaxf(mx, __shfl_xor(mx, off, 32));
    float sum = 0.0f;
    for (int c = lane; c < 2048; c += 32) {
      float e = __expf(Srow[c] - mx);
      Srow[c] = e;
      sum += e;
    }
#pragma unroll
    for (int off = 16; off; off >>= 1) sum += __shfl_xor(sum, off, 32);
    float inv = 1.0f / sum;
    for (int c = lane; c < 2048; c += 32) Srow[c] *= inv;
  }
  __syncthreads();

  // ---- O = P V : wave owns its 256-key slice, partials reduced via ds_add_f32 ----
  v8f o[4] = {};
  for (int sc = 0; sc < 8; ++sc) {
    int sb = cw + sc * 32;
    v16bf pa;
    {
      const float4* p4 = reinterpret_cast<const float4*>(
          S + (lane & 15) * SLDA + sb + ((lane >> 4) << 3));
      float4 x0 = p4[0], x1 = p4[1];   // K +0..7
      float4 y0 = p4[4], y1 = p4[5];   // K +16..23
      pa[0] = (bf16)x0.x; pa[1] = (bf16)x0.y; pa[2] = (bf16)x0.z; pa[3] = (bf16)x0.w;
      pa[4] = (bf16)x1.x; pa[5] = (bf16)x1.y; pa[6] = (bf16)x1.z; pa[7] = (bf16)x1.w;
      pa[8]  = (bf16)y0.x; pa[9]  = (bf16)y0.y; pa[10] = (bf16)y0.z; pa[11] = (bf16)y0.w;
      pa[12] = (bf16)y1.x; pa[13] = (bf16)y1.y; pa[14] = (bf16)y1.z; pa[15] = (bf16)y1.w;
    }
#pragma unroll
    for (int tn = 0; tn < 4; ++tn)
      o[tn] = wmma_bf16(pa, load_b_frag(Vh, 2048, tn * 16, sb), o[tn]);
  }
#pragma unroll
  for (int tn = 0; tn < 4; ++tn)
#pragma unroll
    for (int r = 0; r < 8; ++r)
      atomicAdd(&Oacc[(rb + r) * 64 + tn * 16 + (lane & 15)], o[tn][r]);
  __syncthreads();

  for (int i = tid; i < 1024; i += 256) {
    int r = i >> 6, dd = i & 63;
    Yb[(q0 + r) * 2048 + h * 64 + dd] = (bf16)Oacc[i];
  }
}

// ---------------- GEMM3: out = Y @ w_proj (f32 out), 64x64 register tile ----------------
__global__ __launch_bounds__(256) void gemm_proj(const bf16* __restrict__ yb,
                                                 const bf16* __restrict__ wT,
                                                 float* __restrict__ out) {
  int wid = blockIdx.x * 8 + (threadIdx.x >> 5);
  int m0 = (wid / 32) * 64;
  int n0 = (wid % 32) * 64;
  v8f acc[4][4] = {};
  for (int kb = 0; kb < 2048; kb += 32) {
    v16bf a[4];
#pragma unroll
    for (int mi = 0; mi < 4; ++mi) a[mi] = load_a_frag(yb, 2048, m0 + 16 * mi, kb);
#pragma unroll
    for (int j = 0; j < 4; ++j) {
      v16bf b = load_b_frag(wT, 2048, n0 + 16 * j, kb);
#pragma unroll
      for (int mi = 0; mi < 4; ++mi) acc[mi][j] = wmma_bf16(a[mi], b, acc[mi][j]);
    }
  }
  int lane = threadIdx.x & 31;
  int rb = (lane >> 4) * 8;
#pragma unroll
  for (int mi = 0; mi < 4; ++mi)
#pragma unroll
    for (int j = 0; j < 4; ++j) {
      int n = n0 + 16 * j + (lane & 15);
#pragma unroll
      for (int r = 0; r < 8; ++r)
        out[(m0 + 16 * mi + rb + r) * 2048 + n] = acc[mi][j][r];
    }
}

// ---------------- host ----------------

extern "C" void kernel_launch(void* const* d_in, const int* in_sizes, int n_in,
                              void* d_out, int out_size, void* d_ws, size_t ws_size,
                              hipStream_t stream) {
  (void)in_sizes; (void)n_in; (void)out_size; (void)ws_size;
  const float* x      = (const float*)d_in[0];   // [2048][2048]
  const float* w_qkv  = (const float*)d_in[1];   // [2048][6144]
  const float* w_proj = (const float*)d_in[2];   // [2048][2048]
  const float* mask   = (const float*)d_in[3];   // [2048][2048]
  float* out = (float*)d_out;

  bf16* xb     = (bf16*)d_ws;                              //  4M elems
  bf16* wqkvT  = xb     + (size_t)4  * 1024 * 1024;        // 12M
  bf16* wprojT = wqkvT  + (size_t)12 * 1024 * 1024;        //  4M
  bf16* Qb     = wprojT + (size_t)4  * 1024 * 1024;        //  4M
  bf16* Kb     = Qb     + (size_t)4  * 1024 * 1024;        //  4M
  bf16* Vt     = Kb     + (size_t)4  * 1024 * 1024;        //  4M
  bf16* Yb     = Vt     + (size_t)4  * 1024 * 1024;        //  4M  (72 MiB total)

  cvt_f32_bf16x4<<<(1024 * 1024) / 256, 256, 0, stream>>>(
      (const float4*)x, (v4bf*)xb, 1024 * 1024);
  dim3 tb(32, 8);
  transpose_cvt<<<dim3(192, 64), tb, 0, stream>>>(w_qkv, wqkvT, 2048, 6144);
  transpose_cvt<<<dim3(64, 64), tb, 0, stream>>>(w_proj, wprojT, 2048, 2048);

  gemm_qkv_rope<<<(32 * 96) / 8, 256, 0, stream>>>(xb, wqkvT, Qb, Kb, Vt);

  size_t shmem = (size_t)(16 * SLDA + 1024) * sizeof(float);  // ~132 KB < 320 KB/WGP
  hipFuncSetAttribute(reinterpret_cast<const void*>(attn),
                      hipFuncAttributeMaxDynamicSharedMemorySize, (int)shmem);
  attn<<<32 * 128, 256, shmem, stream>>>(Qb, Kb, Vt, mask, Yb);

  gemm_proj<<<(32 * 32) / 8, 256, 0, stream>>>(Yb, wprojT, out);
}
